// Attention_59150289600879
// MI455X (gfx1250) — matrix operand order repaired and checked by reading, not compiled, not run on hardware
//
#include <hip/hip_runtime.h>
#include <hip/hip_bf16.h>
#include <math.h>

typedef __bf16 v16bf __attribute__((ext_vector_type(16)));
typedef __bf16 bf16x8 __attribute__((ext_vector_type(8)));
typedef float  v8f   __attribute__((ext_vector_type(8)));

#define DIM   1024
#define NHEAD 16
#define HDIM  64
#define SEQ   2048
#define BATCH 2
#define MROWS (BATCH * SEQ)   /* 4096 */
#define SCALE 0.125f          /* 1/sqrt(64) */

static __device__ __forceinline__ v16bf load16(const __bf16* p0, const __bf16* p1) {
  union { bf16x8 h[2]; v16bf v; } u;
  u.h[0] = *(const bf16x8*)p0;
  u.h[1] = *(const bf16x8*)p1;
  return u.v;
}

static __device__ __forceinline__ v8f wmma_bf16(v16bf a, v16bf b, v8f c) {
  return __builtin_amdgcn_wmma_f32_16x16x32_bf16(false, a, false, b, (short)0, c,
                                                 false, false);
}

__global__ void cvt_f32_bf16(const float* __restrict__ in,
                             __bf16* __restrict__ out, int n) {
  int i = blockIdx.x * blockDim.x + threadIdx.x;
  if (i < n) out[i] = (__bf16)in[i];
}

// Wave computes a 32x64 output tile at (rowBase, colBase): acc[2][4] of 16x16.
// A is [M,K] row-major bf16; W is [N,K] row-major bf16 (i.e. B = W^T).
static __device__ __forceinline__ void gemm_core(const __bf16* __restrict__ A,
                                                 const __bf16* __restrict__ W,
                                                 int K, int rowBase, int colBase,
                                                 int l16, int half,
                                                 v8f (&acc)[2][4]) {
  for (int kk = 0; kk < K; kk += 32) {
    v16bf a[2];
#pragma unroll
    for (int i = 0; i < 2; ++i) {
      const __bf16* base = A + (size_t)(rowBase + i * 16 + l16) * K + kk;
      a[i] = load16(base + half * 8, base + 16 + half * 8);
    }
#pragma unroll
    for (int j = 0; j < 4; ++j) {
      const __bf16* base =
          W + (size_t)(colBase + j * 16 + l16) * K + kk;
      v16bf b = load16(base + half * 8, base + 16 + half * 8);
#pragma unroll
      for (int i = 0; i < 2; ++i) acc[i][j] = wmma_bf16(a[i], b, acc[i][j]);
    }
  }
}

// QKV GEMM: [4096,1024] x [1024,3072] + bias -> scatter into q/k/vT head layouts.
__global__ void __launch_bounds__(128)
gemm_qkv_kernel(const __bf16* __restrict__ A, const __bf16* __restrict__ W,
                const float* __restrict__ bias, __bf16* __restrict__ qo,
                __bf16* __restrict__ ko, __bf16* __restrict__ vT) {
  int lane = threadIdx.x & 31, w = threadIdx.x >> 5;
  int l16 = lane & 15, half = lane >> 4;
  int rowBase = blockIdx.y * 64 + (w >> 1) * 32;
  int colBase = blockIdx.x * 128 + (w & 1) * 64;
  v8f acc[2][4] = {};
  gemm_core(A, W, DIM, rowBase, colBase, l16, half, acc);
#pragma unroll
  for (int i = 0; i < 2; ++i) {
#pragma unroll
    for (int j = 0; j < 4; ++j) {
      int n = colBase + j * 16 + l16;
      int which = n >> 10;        // 0=q 1=k 2=v
      int within = n & 1023;
      int head = within >> 6;
      int d = within & 63;
      float bv = bias[n];
#pragma unroll
      for (int r = 0; r < 8; ++r) {
        int m = rowBase + i * 16 + r + half * 8;
        int b = m >> 11;          // /SEQ
        int seq = m & (SEQ - 1);
        int bh = b * NHEAD + head;
        float val = acc[i][j][r] + bv;
        if (which == 0)
          qo[((size_t)bh * SEQ + seq) * HDIM + d] = (__bf16)(val * SCALE);
        else if (which == 1)
          ko[((size_t)bh * SEQ + seq) * HDIM + d] = (__bf16)val;
        else
          vT[((size_t)bh * HDIM + d) * SEQ + seq] = (__bf16)val;
      }
    }
  }
}

// Flash attention: one wave per (bh, 32-query tile). Online softmax over 32-key
// blocks. Each K/V fragment is loaded once and reused for both 16-row query
// sub-tiles (halves K/V read traffic vs a 16-query tile).
// q pre-scaled by SCALE, vT stored [bh, d, seq].
__global__ void __launch_bounds__(32)
attn_flash_kernel(const __bf16* __restrict__ q, const __bf16* __restrict__ k,
                  const __bf16* __restrict__ vT, __bf16* __restrict__ ao) {
  __shared__ __bf16 pbuf[2 * 16 * 40] __attribute__((aligned(16)));
  int lane = threadIdx.x;
  int l16 = lane & 15, half = lane >> 4;
  int bh = blockIdx.y;
  int b = bh >> 4, h = bh & 15;
  int qBase = blockIdx.x * 32;
  const __bf16* qh = q + (size_t)bh * SEQ * HDIM;
  const __bf16* kh = k + (size_t)bh * SEQ * HDIM;
  const __bf16* vh = vT + (size_t)bh * HDIM * SEQ;

  v16bf aQ[2][2];   // [query sub-tile][d-half]
#pragma unroll
  for (int qt = 0; qt < 2; ++qt) {
#pragma unroll
    for (int s = 0; s < 2; ++s) {
      const __bf16* base = qh + (size_t)(qBase + qt * 16 + l16) * HDIM + s * 32;
      aQ[qt][s] = load16(base + half * 8, base + 16 + half * 8);
    }
  }

  v8f o[2][4] = {};
  float mrow[2][8], lrow[2][8];
#pragma unroll
  for (int qt = 0; qt < 2; ++qt)
#pragma unroll
    for (int r = 0; r < 8; ++r) { mrow[qt][r] = -3.0e38f; lrow[qt][r] = 0.0f; }

  for (int kb = 0; kb < SEQ; kb += 32) {
    // S = Q K^T for 32 queries x 32 keys (four 16x16 C tiles); K frags shared.
    v8f st[2][2];
#pragma unroll
    for (int g = 0; g < 2; ++g) {
      v8f c0 = {}, c1 = {};
#pragma unroll
      for (int s = 0; s < 2; ++s) {
        const __bf16* base =
            kh + (size_t)(kb + g * 16 + l16) * HDIM + s * 32;
        v16bf bK = load16(base + half * 8, base + 16 + half * 8);
        c0 = wmma_bf16(aQ[0][s], bK, c0);
        c1 = wmma_bf16(aQ[1][s], bK, c1);
      }
      st[0][g] = c0;
      st[1][g] = c1;
    }
    // online softmax per query sub-tile; rows reduce across 16-lane groups
    float alpha[2][8];
#pragma unroll
    for (int qt = 0; qt < 2; ++qt) {
#pragma unroll
      for (int r = 0; r < 8; ++r) {
        float mx = fmaxf(st[qt][0][r], st[qt][1][r]);
        mx = fmaxf(mx, __shfl_xor(mx, 1));
        mx = fmaxf(mx, __shfl_xor(mx, 2));
        mx = fmaxf(mx, __shfl_xor(mx, 4));
        mx = fmaxf(mx, __shfl_xor(mx, 8));
        float mn = fmaxf(mrow[qt][r], mx);
        alpha[qt][r] = __expf(mrow[qt][r] - mn);
        mrow[qt][r] = mn;
        float p0 = __expf(st[qt][0][r] - mn);
        float p1 = __expf(st[qt][1][r] - mn);
        st[qt][0][r] = p0;
        st[qt][1][r] = p1;
        float rs = p0 + p1;
        rs += __shfl_xor(rs, 1);
        rs += __shfl_xor(rs, 2);
        rs += __shfl_xor(rs, 4);
        rs += __shfl_xor(rs, 8);
        lrow[qt][r] = lrow[qt][r] * alpha[qt][r] + rs;
      }
#pragma unroll
      for (int t = 0; t < 4; ++t)
#pragma unroll
        for (int r = 0; r < 8; ++r) o[qt][t][r] *= alpha[qt][r];
      // P -> LDS (row-major 16x32, row stride 40 elems: bank spread + 16B align)
#pragma unroll
      for (int g = 0; g < 2; ++g)
#pragma unroll
        for (int r = 0; r < 8; ++r)
          pbuf[qt * 640 + (r + half * 8) * 40 + g * 16 + l16] =
              (__bf16)st[qt][g][r];
    }
    __syncthreads();
    v16bf aP[2];
#pragma unroll
    for (int qt = 0; qt < 2; ++qt)
      aP[qt] = load16(&pbuf[qt * 640 + l16 * 40 + half * 8],
                      &pbuf[qt * 640 + l16 * 40 + 16 + half * 8]);
    // O += P V ; V frags shared between the two query sub-tiles
#pragma unroll
    for (int t = 0; t < 4; ++t) {
      const __bf16* base = vh + (size_t)(t * 16 + l16) * SEQ + kb;
      v16bf bV = load16(base + half * 8, base + 16 + half * 8);
      o[0][t] = wmma_bf16(aP[0], bV, o[0][t]);
      o[1][t] = wmma_bf16(aP[1], bV, o[1][t]);
    }
    __syncthreads();
  }
  // finalize: divide by l, write back to [B, N, DIM] interleaved-head bf16
#pragma unroll
  for (int qt = 0; qt < 2; ++qt) {
#pragma unroll
    for (int r = 0; r < 8; ++r) {
      float inv = 1.0f / lrow[qt][r];
      int seq = qBase + qt * 16 + r + half * 8;
      __bf16* dst = ao + ((size_t)(b * SEQ + seq)) * DIM + h * HDIM;
#pragma unroll
      for (int t = 0; t < 4; ++t) dst[t * 16 + l16] = (__bf16)(o[qt][t][r] * inv);
    }
  }
}

// Projection GEMM: [4096,1024] x [1024,1024] + bias -> fp32 out.
__global__ void __launch_bounds__(128)
gemm_proj_kernel(const __bf16* __restrict__ A, const __bf16* __restrict__ W,
                 const float* __restrict__ bias, float* __restrict__ out) {
  int lane = threadIdx.x & 31, w = threadIdx.x >> 5;
  int l16 = lane & 15, half = lane >> 4;
  int rowBase = blockIdx.y * 64 + (w >> 1) * 32;
  int colBase = blockIdx.x * 128 + (w & 1) * 64;
  v8f acc[2][4] = {};
  gemm_core(A, W, DIM, rowBase, colBase, l16, half, acc);
#pragma unroll
  for (int i = 0; i < 2; ++i) {
#pragma unroll
    for (int j = 0; j < 4; ++j) {
      int n = colBase + j * 16 + l16;
      float bv = bias[n];
#pragma unroll
      for (int r = 0; r < 8; ++r) {
        int m = rowBase + i * 16 + r + half * 8;
        out[(size_t)m * DIM + n] = acc[i][j][r] + bv;
      }
    }
  }
}

extern "C" void kernel_launch(void* const* d_in, const int* in_sizes, int n_in,
                              void* d_out, int out_size, void* d_ws,
                              size_t ws_size, hipStream_t stream) {
  (void)in_sizes; (void)n_in; (void)out_size; (void)ws_size;
  const float* x      = (const float*)d_in[0];
  const float* qkv_w  = (const float*)d_in[1];
  const float* qkv_b  = (const float*)d_in[2];
  const float* proj_w = (const float*)d_in[3];
  const float* proj_b = (const float*)d_in[4];
  float* out = (float*)d_out;

  __bf16* p = (__bf16*)d_ws;
  __bf16* xh   = p; p += (size_t)MROWS * DIM;      // 4M elems
  __bf16* wqkv = p; p += (size_t)3 * DIM * DIM;    // 3M
  __bf16* wp   = p; p += (size_t)DIM * DIM;        // 1M
  __bf16* qb   = p; p += (size_t)MROWS * DIM;      // 4M
  __bf16* kb   = p; p += (size_t)MROWS * DIM;      // 4M
  __bf16* vT   = p; p += (size_t)MROWS * DIM;      // 4M
  __bf16* ao   = p;                                // 4M  (~48 MB total)

  int n1 = MROWS * DIM, n2 = 3 * DIM * DIM, n3 = DIM * DIM;
  cvt_f32_bf16<<<(n1 + 255) / 256, 256, 0, stream>>>(x, xh, n1);
  cvt_f32_bf16<<<(n2 + 255) / 256, 256, 0, stream>>>(qkv_w, wqkv, n2);
  cvt_f32_bf16<<<(n3 + 255) / 256, 256, 0, stream>>>(proj_w, wp, n3);

  gemm_qkv_kernel<<<dim3(3 * DIM / 128, MROWS / 64), 128, 0, stream>>>(
      xh, wqkv, qkv_b, qb, kb, vT);

  attn_flash_kernel<<<dim3(SEQ / 32, BATCH * NHEAD), 32, 0, stream>>>(qb, kb, vT,
                                                                      ao);

  gemm_proj_kernel<<<dim3(DIM / 128, MROWS / 64), 128, 0, stream>>>(ao, wp,
                                                                    proj_b, out);
}
